// HGRN2Cell_65687229825432
// MI455X (gfx1250) — matrix operand order repaired and checked by reading, not compiled, not run on hardware
//
#include <hip/hip_runtime.h>
#include <hip/hip_bf16.h>

#define D_MODEL 2048
#define EXPAND  8
#define LB      (1.0f/32.0f)

// Block tile: BM=128 (8 waves x 16 rows), BN=32 (2 n-tiles), BK=64 (2 WMMA K-steps)
#define BM 128
#define BN 32
#define BK 64

typedef __attribute__((ext_vector_type(16))) __bf16        v16bf;
typedef __attribute__((ext_vector_type(8)))  __bf16        v8bf;
typedef __attribute__((ext_vector_type(4)))  __bf16        v4bf;
typedef __attribute__((ext_vector_type(8)))  float         v8f;
typedef __attribute__((ext_vector_type(4)))  float         v4f;
typedef __attribute__((ext_vector_type(4)))  unsigned int  v4u;
typedef __attribute__((ext_vector_type(8)))  int           v8i;
typedef __attribute__((ext_vector_type(4)))  int           v4i;

// ---- TDM availability / arity detection (compile-time) ----
#ifdef __has_builtin
#  if __has_builtin(__builtin_amdgcn_tensor_load_to_lds)
#    define HAS_TDM 1
#  else
#    define HAS_TDM 0
#  endif
#else
#  define HAS_TDM 0
#endif
#if defined(__has_include)
#  if __has_include(<hip/amd_detail/amd_gfx1250_TDM.h>)
#    define TDM_6ARG 1
#  else
#    define TDM_6ARG 0
#  endif
#else
#  define TDM_6ARG 0
#endif

__device__ __forceinline__ v8bf cvt8(const float* __restrict__ p) {
    v4f a = *(const v4f*)p;
    v4f b = *(const v4f*)(p + 4);
    v8bf r;
    r[0] = (__bf16)a[0]; r[1] = (__bf16)a[1]; r[2] = (__bf16)a[2]; r[3] = (__bf16)a[3];
    r[4] = (__bf16)b[0]; r[5] = (__bf16)b[1]; r[6] = (__bf16)b[2]; r[7] = (__bf16)b[3];
    return r;
}

__device__ __forceinline__ float sigmoid_f(float z) {
    return 1.0f / (1.0f + __expf(-z));
}

// Shared fused epilogue: bias + gates + h recurrence + E-reduction + both outputs.
// C/D layout: element i -> (m = m_tile + half*8 + i, n = n_tile + (lane&15)).
__device__ __forceinline__ void hgrn2_epilogue(
    const v8f* accF, const v8f* accU, const v8f* accO,
    const float* __restrict__ h,
    const float* __restrict__ bf_, const float* __restrict__ bu_,
    const float* __restrict__ bo_,
    float* __restrict__ out, float* __restrict__ hnew,
    int m_blk, int n_blk, int wave, int half, int nl)
{
    const float inv_sqrtE = 0.35355339059327373f; // 1/sqrt(8)
    #pragma unroll
    for (int nt = 0; nt < 2; ++nt) {
        const int   n   = n_blk + nt * 16 + nl;
        const float bfv = bf_[n];
        const float buv = bu_[n];
        const float bov = bo_[n];
        #pragma unroll
        for (int i = 0; i < 8; ++i) {
            const int m = m_blk + wave * 16 + half * 8 + i;
            const float zf = accF[nt][i] + bfv;
            const float zu = accU[nt][i] + buv;
            const float zo = accO[nt][i] + bov;
            const float f = LB + (1.0f - LB) * sigmoid_f(zf);
            const float u = zu * sigmoid_f(zu);   // silu
            const float o = sigmoid_f(zo);

            const size_t hoff = ((size_t)m * D_MODEL + n) * EXPAND;
            v4f h0 = *(const v4f*)(h + hoff);
            v4f h1 = *(const v4f*)(h + hoff + 4);
            const float fu = (1.0f - f) * u;
            v4f hn0, hn1;
            #pragma unroll
            for (int c = 0; c < 4; ++c) {
                hn0[c] = f * h0[c] + fu;
                hn1[c] = f * h1[c] + fu;
            }
            *(v4f*)(hnew + hoff)     = hn0;
            *(v4f*)(hnew + hoff + 4) = hn1;
            const float y = (hn0[0] + hn0[1] + hn0[2] + hn0[3] +
                             hn1[0] + hn1[1] + hn1[2] + hn1[3]) * inv_sqrtE;
            out[(size_t)m * D_MODEL + n] = o * y;
        }
    }
}

// ---- fp32 -> bf16 pre-conversion (pure bandwidth) ----
__global__ __launch_bounds__(256) void cvt_f32_to_bf16(
    const float* __restrict__ src, unsigned short* __restrict__ dst, int n4)
{
    const int i = blockIdx.x * blockDim.x + threadIdx.x;
    if (i >= n4) return;
    v4f v = *(const v4f*)(src + (size_t)i * 4);
    v4bf r;
    r[0] = (__bf16)v[0]; r[1] = (__bf16)v[1]; r[2] = (__bf16)v[2]; r[3] = (__bf16)v[3];
    *(v4bf*)(dst + (size_t)i * 4) = r;
}

#if HAS_TDM
// Issue one TDM 2D-tile load: 32 rows x 64 cols of bf16 from a [2048 x 2048]
// bf16 matrix (row stride 2048) into a packed 32x64 LDS tile.
// D# per CDNA5 ISA §8.3-8.4 (groups 0/1; 2D so groups 2/3 zero).
__device__ __forceinline__ void tdm_load_tile(unsigned lds_addr,
                                              const __bf16* __restrict__ gptr)
{
    const unsigned long long ga = (unsigned long long)(size_t)gptr;
    v4u g0;
    g0[0] = 1u;                                        // count=1, user mode
    g0[1] = lds_addr;                                  // lds_addr
    g0[2] = (unsigned)(ga & 0xFFFFFFFFu);              // global_addr[31:0]
    g0[3] = (unsigned)((ga >> 32) & 0x01FFFFFFu)       // global_addr[56:32]
          | (2u << 30);                                // type=2 ("image")
    v8i g1;
    g1[0] = (int)(1u << 16);                           // wg_mask=0, data_size=2B
    g1[1] = (int)(((unsigned)D_MODEL & 0xFFFFu) << 16);// tensor_dim0[15:0]
    g1[2] = (int)(((unsigned)D_MODEL & 0xFFFFu) << 16);// dim0[31:16]=0 | dim1[15:0]
    g1[3] = (int)(64u << 16);                          // dim1[31:16]=0 | tile_dim0=64
    g1[4] = 32;                                        // tile_dim1=32, tile_dim2=0
    g1[5] = D_MODEL;                                   // tensor_dim0_stride[31:0]
    g1[6] = 0;                                         // stride hi / dim1_stride lo
    g1[7] = 0;
    v4i z4 = {0, 0, 0, 0};
#if TDM_6ARG
    v8i z8 = {0, 0, 0, 0, 0, 0, 0, 0};
    __builtin_amdgcn_tensor_load_to_lds(g0, g1, z4, z4, z8, 0);
#else
    __builtin_amdgcn_tensor_load_to_lds(g0, g1, z4, z4, 0);
#endif
}
#endif // HAS_TDM

// ---------------------------------------------------------------------------
// Fast path: x and W pre-converted to bf16 in workspace. W tiles staged to LDS
// by the Tensor Data Mover (wave 0 issues, TENSORcnt + barrier sync), double
// buffered; A fragments are straight b128 loads. 12 WMMA per wave per K-tile.
// ---------------------------------------------------------------------------
__global__ __launch_bounds__(256) void hgrn2_fused_bf16(
    const unsigned short* __restrict__ xb_u,  // [4096, 2048] bf16
    const float* __restrict__ h,
    const unsigned short* __restrict__ wb_u,  // [3][2048][2048] bf16
    const float* __restrict__ bf_,
    const float* __restrict__ bu_,
    const float* __restrict__ bo_,
    float* __restrict__ out,
    float* __restrict__ hnew)
{
    __shared__ __bf16 ldsB[2][3][BN][BK];     // double-buffered, 24 KB

    const __bf16* xb = (const __bf16*)xb_u;
    const __bf16* Wb = (const __bf16*)wb_u;

    const int tid   = threadIdx.x;
    const int lane  = tid & 31;
    const int wave  = tid >> 5;
    const int half  = lane >> 4;
    const int nl    = lane & 15;
    const int m_blk = blockIdx.x * BM;
    const int n_blk = blockIdx.y * BN;

    v8f accF[2] = {{}, {}};
    v8f accU[2] = {{}, {}};
    v8f accO[2] = {{}, {}};

    const int     arow = m_blk + wave * 16 + nl;
    const __bf16* ap   = xb + (size_t)arow * D_MODEL;

#if HAS_TDM
    const unsigned lds0      = (unsigned)(size_t)&ldsB[0][0][0][0];
    const unsigned tileBytes = BN * BK * 2;   // 4096
    if (wave == 0) {
        #pragma unroll
        for (int p = 0; p < 3; ++p)
            tdm_load_tile(lds0 + p * tileBytes,
                          Wb + (size_t)p * D_MODEL * D_MODEL + (size_t)n_blk * D_MODEL);
    }
#else
    const int srow = tid >> 3;
    const int scol = (tid & 7) * 8;
    #pragma unroll
    for (int p = 0; p < 3; ++p) {
        v8bf wv = *(const v8bf*)(Wb + (size_t)p * D_MODEL * D_MODEL +
                                 (size_t)(n_blk + srow) * D_MODEL + scol);
        *(v8bf*)&ldsB[0][p][srow][scol] = wv;
    }
#endif

    int buf = 0;
    for (int kb = 0; kb < D_MODEL; kb += BK) {
        const bool more = (kb + BK) < D_MODEL;
#if HAS_TDM
        if (wave == 0) {
            if (more) {
                #pragma unroll
                for (int p = 0; p < 3; ++p)
                    tdm_load_tile(lds0 + ((buf ^ 1) * 3 + p) * tileBytes,
                                  Wb + (size_t)p * D_MODEL * D_MODEL +
                                  (size_t)n_blk * D_MODEL + kb + BK);
                __builtin_amdgcn_s_wait_tensorcnt(3);  // current buffer's 3 done
            } else {
                __builtin_amdgcn_s_wait_tensorcnt(0);
            }
        }
#else
        if (more) {
            #pragma unroll
            for (int p = 0; p < 3; ++p) {
                v8bf wv = *(const v8bf*)(Wb + (size_t)p * D_MODEL * D_MODEL +
                                         (size_t)(n_blk + srow) * D_MODEL + kb + BK + scol);
                *(v8bf*)&ldsB[buf ^ 1][p][srow][scol] = wv;
            }
        }
#endif
        __syncthreads();   // current buffer valid for all waves

        #pragma unroll
        for (int ks = 0; ks < BK; ks += 32) {
            // A fragment (16x32 bf16): lane chunks at K = 16*half and +8
            v8bf alo = *(const v8bf*)(ap + kb + ks + half * 16);
            v8bf ahi = *(const v8bf*)(ap + kb + ks + half * 16 + 8);
            v16bf afrag = __builtin_shufflevector(alo, ahi,
                0,1,2,3,4,5,6,7,8,9,10,11,12,13,14,15);

            #pragma unroll
            for (int nt = 0; nt < 2; ++nt) {
                const int br = nt * 16 + nl;
                v16bf bF = *(const v16bf*)&ldsB[buf][0][br][ks + half * 16];
                v16bf bU = *(const v16bf*)&ldsB[buf][1][br][ks + half * 16];
                v16bf bO = *(const v16bf*)&ldsB[buf][2][br][ks + half * 16];

                accF[nt] = __builtin_amdgcn_wmma_f32_16x16x32_bf16(
                    false, afrag, false, bF, (short)0, accF[nt], false, false);
                accU[nt] = __builtin_amdgcn_wmma_f32_16x16x32_bf16(
                    false, afrag, false, bU, (short)0, accU[nt], false, false);
                accO[nt] = __builtin_amdgcn_wmma_f32_16x16x32_bf16(
                    false, afrag, false, bO, (short)0, accO[nt], false, false);
            }
        }
        __syncthreads();   // all reads of current buffer done before it is refilled
        buf ^= 1;
    }

    hgrn2_epilogue(accF, accU, accO, h, bf_, bu_, bo_, out, hnew,
                   m_blk, n_blk, wave, half, nl);
}

// ---------------------------------------------------------------------------
// Fallback (no workspace): convert-in-loop version.
// ---------------------------------------------------------------------------
__global__ __launch_bounds__(256) void hgrn2_fused(
    const float* __restrict__ x,
    const float* __restrict__ h,
    const float* __restrict__ Wf, const float* __restrict__ bf_,
    const float* __restrict__ Wu, const float* __restrict__ bu_,
    const float* __restrict__ Wo, const float* __restrict__ bo_,
    float* __restrict__ out, float* __restrict__ hnew)
{
    __shared__ __bf16 ldsB[3][BN][BK];   // 12 KB

    const int tid   = threadIdx.x;
    const int lane  = tid & 31;
    const int wave  = tid >> 5;
    const int half  = lane >> 4;
    const int nl    = lane & 15;
    const int m_blk = blockIdx.x * BM;
    const int n_blk = blockIdx.y * BN;

    const float* Wmat[3] = {Wf, Wu, Wo};

    v8f accF[2] = {{}, {}};
    v8f accU[2] = {{}, {}};
    v8f accO[2] = {{}, {}};

    const int srow = tid >> 3;
    const int scol = (tid & 7) * 8;

    const int    arow = m_blk + wave * 16 + nl;
    const float* aptr = x + (size_t)arow * D_MODEL;

    for (int kb = 0; kb < D_MODEL; kb += BK) {
        if (kb + BK < D_MODEL) __builtin_prefetch(aptr + kb + BK, 0, 0);

        #pragma unroll
        for (int p = 0; p < 3; ++p) {
            const float* wp = Wmat[p] + (size_t)(n_blk + srow) * D_MODEL + kb + scol;
            v8bf wv = cvt8(wp);
            *(v8bf*)&ldsB[p][srow][scol] = wv;
        }
        __syncthreads();

        #pragma unroll
        for (int ks = 0; ks < BK; ks += 32) {
            v8bf alo = cvt8(aptr + kb + ks + half * 16);
            v8bf ahi = cvt8(aptr + kb + ks + half * 16 + 8);
            v16bf afrag = __builtin_shufflevector(alo, ahi,
                0,1,2,3,4,5,6,7,8,9,10,11,12,13,14,15);

            #pragma unroll
            for (int nt = 0; nt < 2; ++nt) {
                const int br = nt * 16 + nl;
                v16bf bF = *(const v16bf*)&ldsB[0][br][ks + half * 16];
                v16bf bU = *(const v16bf*)&ldsB[1][br][ks + half * 16];
                v16bf bO = *(const v16bf*)&ldsB[2][br][ks + half * 16];

                accF[nt] = __builtin_amdgcn_wmma_f32_16x16x32_bf16(
                    false, afrag, false, bF, (short)0, accF[nt], false, false);
                accU[nt] = __builtin_amdgcn_wmma_f32_16x16x32_bf16(
                    false, afrag, false, bU, (short)0, accU[nt], false, false);
                accO[nt] = __builtin_amdgcn_wmma_f32_16x16x32_bf16(
                    false, afrag, false, bO, (short)0, accO[nt], false, false);
            }
        }
        __syncthreads();
    }

    hgrn2_epilogue(accF, accU, accO, h, bf_, bu_, bo_, out, hnew,
                   m_blk, n_blk, wave, half, nl);
}

extern "C" void kernel_launch(void* const* d_in, const int* in_sizes, int n_in,
                              void* d_out, int out_size, void* d_ws, size_t ws_size,
                              hipStream_t stream) {
    const float* x   = (const float*)d_in[0];
    const float* h   = (const float*)d_in[1];
    const float* Wf  = (const float*)d_in[2];
    const float* bf_ = (const float*)d_in[3];
    const float* Wu  = (const float*)d_in[4];
    const float* bu_ = (const float*)d_in[5];
    const float* Wo  = (const float*)d_in[6];
    const float* bo_ = (const float*)d_in[7];

    const int Bsz = in_sizes[0] / D_MODEL;   // 4096

    float* out  = (float*)d_out;
    float* hnew = out + (size_t)Bsz * D_MODEL;

    const size_t elemsX = (size_t)Bsz * D_MODEL;
    const size_t elemsW = (size_t)D_MODEL * D_MODEL;
    const size_t need   = (elemsX + 3 * elemsW) * 2;   // bf16 bytes

    dim3 grid(Bsz / BM, D_MODEL / BN);

    if (ws_size >= need) {
        unsigned short* xb = (unsigned short*)d_ws;
        unsigned short* wb = xb + elemsX;              // [3][D][D]

        const int n4x = (int)(elemsX / 4);
        const int n4w = (int)(elemsW / 4);
        cvt_f32_to_bf16<<<(n4x + 255) / 256, 256, 0, stream>>>(x,  xb, n4x);
        cvt_f32_to_bf16<<<(n4w + 255) / 256, 256, 0, stream>>>(Wf, wb,              n4w);
        cvt_f32_to_bf16<<<(n4w + 255) / 256, 256, 0, stream>>>(Wu, wb + elemsW,     n4w);
        cvt_f32_to_bf16<<<(n4w + 255) / 256, 256, 0, stream>>>(Wo, wb + 2 * elemsW, n4w);

        hgrn2_fused_bf16<<<grid, 256, 0, stream>>>(xb, h, wb, bf_, bu_, bo_, out, hnew);
    } else {
        hgrn2_fused<<<grid, 256, 0, stream>>>(x, h, Wf, bf_, Wu, bu_, Wo, bo_, out, hnew);
    }
}
